// LightningIndexer_63909113365062
// MI455X (gfx1250) — matrix operand order repaired and checked by reading, not compiled, not run on hardware
//
#include <hip/hip_runtime.h>
#include <stdint.h>

typedef __attribute__((ext_vector_type(16))) __bf16 v16bf;
typedef __attribute__((ext_vector_type(8)))  float  v8f;

#define H_    16
#define TQ_   4096
#define TKV_  4096
#define DH_   128
#define DM_   2048   /* H*DH  */
#define NI_   1024   /* H*IDX */
#define IDX_  64
#define NSEL_ 512
#define SCALE_ 0.125f /* 1/sqrt(64) */

// LDS layout for score_topk (dynamic shared)
#define SC_BYTES   ((size_t)16 * TKV_ * 4)            // 262144: score strip
#define KEY_OFF    (SC_BYTES)                         // 512 u64 keys
#define CNT_OFF    (KEY_OFF + (size_t)NSEL_ * 8)      // counters
#define SB_OFF     (CNT_OFF + 64)                     // double-buffered K panel
#define SB_BYTES   ((size_t)2 * 64 * IDX_ * 2)        // 2 x 8KB
#define LDS_TOTAL  (SB_OFF + SB_BYTES)                // ~278KB (< 320KB WGP LDS)

__device__ __forceinline__ uint16_t f32_to_bf16(float f) {
    uint32_t u = __float_as_uint(f);
    u += 0x7FFFu + ((u >> 16) & 1u);                  // round-to-nearest-even
    return (uint16_t)(u >> 16);
}

__device__ __forceinline__ uint32_t ord_f32(float f) {
    // monotonic unsigned mapping of IEEE f32 (larger float -> larger uint)
    uint32_t u = __float_as_uint(f);
    return (u & 0x80000000u) ? ~u : (u | 0x80000000u);
}

// Async global->LDS copy, 16B per lane, tracked by ASYNCcnt (CDNA5).
__device__ __forceinline__ void async_copy16(uint32_t lds_off, const void* g) {
    uint64_t ga = (uint64_t)(uintptr_t)g;
    asm volatile("global_load_async_to_lds_b128 %0, %1, off"
                 :: "v"(lds_off), "v"(ga) : "memory");
}

// ---------------------------------------------------------------------------
// Stage 1a: [B,H,T,DH] f32  ->  [T, H*DH] bf16   (transpose + convert)
// ---------------------------------------------------------------------------
__global__ void pack_states(const float* __restrict__ src, uint16_t* __restrict__ dst) {
    int i = blockIdx.x * 256 + threadIdx.x;           // over T*DM
    if (i >= TQ_ * DM_) return;
    int t = i / DM_, c = i - t * DM_;
    int h = c >> 7, d = c & 127;
    float f = src[((size_t)h * TQ_ + t) * DH_ + d];
    dst[(size_t)t * DM_ + c] = f32_to_bf16(f);
}

// ---------------------------------------------------------------------------
// Stage 1b: W [NI, DM] f32 -> bf16 (same layout)
// ---------------------------------------------------------------------------
__global__ void pack_weight(const float* __restrict__ src, uint16_t* __restrict__ dst) {
    int i = blockIdx.x * 256 + threadIdx.x;           // over NI*DM
    if (i >= NI_ * DM_) return;
    dst[i] = f32_to_bf16(src[i]);
}

// ---------------------------------------------------------------------------
// Stage 2: index projection GEMM  C[T, NI] = A[T, DM] x W^T[DM, NI]
// one wave per 16x16 output tile; K loop in steps of 32 via v_wmma bf16.
// All staging panels (20MB) are L2-resident (192MB), so per-tile re-reads
// hit L2; compute is ~34 GFLOP total -> negligible against WMMA peak.
// Output stored bf16 as [H][T][IDX_].
// ---------------------------------------------------------------------------
union V16U { v16bf v; uint4 u[2]; };

__global__ void __launch_bounds__(128)
proj_gemm(const uint16_t* __restrict__ A, const uint16_t* __restrict__ W,
          uint16_t* __restrict__ out) {
    int lane = threadIdx.x & 31;
    int wave = threadIdx.x >> 5;
    int tile = blockIdx.x * 4 + wave;     // 16384 tiles total
    int n0   = (tile & 63) * 16;          // 64 col tiles
    int t0   = (tile >> 6) * 16;          // 256 row tiles

    int mrow = lane & 15;                 // A row / B col owned by this lane
    int kb8  = (lane >> 4) * 8;           // A: two 8-elem chunks at kb8, kb8+16
    int kb16 = (lane >> 4) * 16;          // unused: B uses kb8, kb8+16 like A

    const uint16_t* arow = A + (size_t)(t0 + mrow) * DM_;
    const uint16_t* brow = W + (size_t)(n0 + mrow) * DM_;

    v8f c = {};
    for (int k0 = 0; k0 < DM_; k0 += 32) {
        __builtin_prefetch(arow + k0 + 256, 0, 0);    // next A panel
        __builtin_prefetch(brow + k0 + 256, 0, 0);    // next B panel
        V16U a, b;
        a.u[0] = *(const uint4*)(arow + k0 + kb8);
        a.u[1] = *(const uint4*)(arow + k0 + kb8 + 16);
        b.u[0] = *(const uint4*)(brow + k0 + kb8);
        b.u[1] = *(const uint4*)(brow + k0 + kb8 + 16);
        c = __builtin_amdgcn_wmma_f32_16x16x32_bf16(false, a.v, false, b.v,
                                                    (short)0, c, false, false);
    }

    // C layout: VGPR r -> row (r + 8*(lane>>4)), col = lane&15
    int N    = lane & 15;
    int mhi  = (lane >> 4) * 8;
    int ncol = n0 + N;
    int h = ncol >> 6, d = ncol & 63;
    for (int r = 0; r < 8; ++r) {
        int row = t0 + r + mhi;
        out[((size_t)h * TQ_ + row) * IDX_ + d] = f32_to_bf16(c[r]);
    }
}

// ---------------------------------------------------------------------------
// Stage 3: fused score + top-k.
// Block = 128 threads (4 waves), one (head, 16-query) tile per block.
// K panels (64 kv rows x 128B = 8KB) are double-buffered into LDS with
// global_load_async_to_lds_b128 (ASYNCcnt), overlapping the copy with the
// previous iteration's WMMA + score writeback. Score strip (16x4096 f32,
// 256KB) lives in the CDNA5 320KB WGP LDS. Top-k: 32-pass radix bisection
// (wave32 shuffle-reduced counts), gather, 512-key bitonic sort.
// ---------------------------------------------------------------------------
__global__ void __launch_bounds__(128)
score_topk(const uint16_t* __restrict__ QI, const uint16_t* __restrict__ KI,
           const float* __restrict__ mask,
           int* __restrict__ out_idx, float* __restrict__ out_val) {
    extern __shared__ unsigned char smem[];
    float*              s_sc  = (float*)smem;                         // [16][4096]
    unsigned long long* s_key = (unsigned long long*)(smem + KEY_OFF);
    int*                s_cnt = (int*)(smem + CNT_OFF);
    unsigned char*      s_B   = smem + SB_OFF;                        // 2 x 8KB

    int tid  = threadIdx.x;
    int lane = tid & 31;
    int wave = tid >> 5;
    int h  = blockIdx.x >> 8;
    int q0 = (blockIdx.x & 255) * 16;

    const uint16_t* qh = QI + (size_t)h * TQ_  * IDX_;
    const uint16_t* kh = KI + (size_t)h * TKV_ * IDX_;

    int mrow = lane & 15;
    int kb8  = (lane >> 4) * 8;
    int kb16 = (lane >> 4) * 16;

    // A tile (16 queries x K=64) is invariant across the KV loop: load once.
    V16U a0, a1;
    const uint16_t* ar = qh + (size_t)(q0 + mrow) * IDX_;
    a0.u[0] = *(const uint4*)(ar + kb8);
    a0.u[1] = *(const uint4*)(ar + kb8 + 16);
    a1.u[0] = *(const uint4*)(ar + 32 + kb8);
    a1.u[1] = *(const uint4*)(ar + 32 + kb8 + 16);

    int N   = lane & 15;
    int mhi = (lane >> 4) * 8;

    // async stage of K panel `it` into buffer it&1 (4 x b128 per thread)
    auto stage = [&](int it) {
        const unsigned char* gp = (const unsigned char*)(kh + (size_t)it * 64 * IDX_);
        uint32_t lb = (uint32_t)(uintptr_t)(s_B + (it & 1) * 8192);
        for (int j = 0; j < 4; ++j) {
            int off = tid * 16 + j * 2048;
            async_copy16(lb + off, gp + off);
        }
    };

    stage(0);
    for (int it = 0; it < TKV_ / 64; ++it) {
        if (it + 1 < TKV_ / 64) {
            stage(it + 1);
            asm volatile("s_wait_asynccnt 0x4" ::: "memory");  // panel `it` landed
        } else {
            asm volatile("s_wait_asynccnt 0x0" ::: "memory");
        }
        __syncthreads();

        const uint16_t* bp = (const uint16_t*)(s_B + (it & 1) * 8192);
        const uint16_t* br = bp + (size_t)(wave * 16 + mrow) * IDX_;
        V16U b0, b1;
        b0.u[0] = *(const uint4*)(br + kb8);
        b0.u[1] = *(const uint4*)(br + kb8 + 16);
        b1.u[0] = *(const uint4*)(br + 32 + kb8);
        b1.u[1] = *(const uint4*)(br + 32 + kb8 + 16);

        v8f c = {};
        c = __builtin_amdgcn_wmma_f32_16x16x32_bf16(false, a0.v, false, b0.v,
                                                    (short)0, c, false, false);
        c = __builtin_amdgcn_wmma_f32_16x16x32_bf16(false, a1.v, false, b1.v,
                                                    (short)0, c, false, false);

        int c0  = it * 64 + wave * 16;
        int col = c0 + N;
        for (int r = 0; r < 8; ++r) {
            int row = r + mhi;
            float v = c[r] * SCALE_ + mask[(size_t)(q0 + row) * TKV_ + col];
            s_sc[row * TKV_ + col] = v;
        }
        __syncthreads();   // panel buffer free for reuse; scores visible
    }

    for (int r = 0; r < 16; ++r) {
        const float* row = s_sc + r * TKV_;

        // ---- radix bisection: T = max threshold with count(ord >= T) >= 512
        uint32_t prefix = 0;
        for (int bit = 31; bit >= 0; --bit) {
            uint32_t cand = prefix | (1u << bit);
            if (tid == 0) s_cnt[0] = 0;
            __syncthreads();
            int local = 0;
            for (int j = tid; j < TKV_; j += 128)
                local += (ord_f32(row[j]) >= cand) ? 1 : 0;
            // wave32 tree reduction; one LDS atomic per wave
            for (int off = 16; off > 0; off >>= 1)
                local += __shfl_xor(local, off, 32);
            if (lane == 0) atomicAdd(&s_cnt[0], local);
            __syncthreads();
            if (s_cnt[0] >= NSEL_) prefix = cand;
            __syncthreads();
        }
        uint32_t T = prefix;

        // ---- gather survivors: strictly greater first, then equals (capped)
        if (tid == 0) s_cnt[1] = 0;
        __syncthreads();
        for (int j = tid; j < TKV_; j += 128) {
            uint32_t m = ord_f32(row[j]);
            if (m > T) {
                int p = atomicAdd(&s_cnt[1], 1);
                s_key[p] = ((unsigned long long)(~m) << 32) | (unsigned)j;
            }
        }
        __syncthreads();
        for (int j = tid; j < TKV_; j += 128) {
            uint32_t m = ord_f32(row[j]);
            if (m == T) {
                int p = atomicAdd(&s_cnt[1], 1);
                if (p < NSEL_)
                    s_key[p] = ((unsigned long long)(~m) << 32) | (unsigned)j;
            }
        }
        __syncthreads();

        // ---- bitonic sort 512 u64 keys ascending
        //      (key = ~ord(val):idx  => descending value, ties by lower index)
        for (int k = 2; k <= NSEL_; k <<= 1) {
            for (int j = k >> 1; j > 0; j >>= 1) {
                for (int t = tid; t < NSEL_; t += 128) {
                    int ixj = t ^ j;
                    if (ixj > t) {
                        unsigned long long x = s_key[t], y = s_key[ixj];
                        bool asc = ((t & k) == 0);
                        if ((x > y) == asc) { s_key[t] = y; s_key[ixj] = x; }
                    }
                }
                __syncthreads();
            }
        }

        // ---- emit
        size_t base = ((size_t)h * TQ_ + (q0 + r)) * NSEL_;
        for (int i = tid; i < NSEL_; i += 128) {
            unsigned long long key = s_key[i];
            int idx = (int)(key & 0xFFFFFFFFull);
            out_idx[base + i] = idx;
            out_val[base + i] = row[idx];
        }
        __syncthreads();
    }
}

// ---------------------------------------------------------------------------
extern "C" void kernel_launch(void* const* d_in, const int* in_sizes, int n_in,
                              void* d_out, int out_size, void* d_ws, size_t ws_size,
                              hipStream_t stream) {
    (void)in_sizes; (void)n_in; (void)out_size; (void)ws_size;
    const float* q    = (const float*)d_in[0];
    const float* k    = (const float*)d_in[1];
    const float* mask = (const float*)d_in[2];
    const float* Wq   = (const float*)d_in[3];
    const float* Wk   = (const float*)d_in[4];

    // workspace layout (bf16 staging, ~56 MB total)
    uint8_t* ws = (uint8_t*)d_ws;
    size_t o = 0;
    uint16_t* qb  = (uint16_t*)(ws + o); o += (size_t)TQ_  * DM_ * 2;  // 16 MB
    uint16_t* kb  = (uint16_t*)(ws + o); o += (size_t)TKV_ * DM_ * 2;  // 16 MB
    uint16_t* wqb = (uint16_t*)(ws + o); o += (size_t)NI_  * DM_ * 2;  //  4 MB
    uint16_t* wkb = (uint16_t*)(ws + o); o += (size_t)NI_  * DM_ * 2;  //  4 MB
    uint16_t* qi  = (uint16_t*)(ws + o); o += (size_t)H_ * TQ_ * IDX_ * 2; // 8 MB
    uint16_t* ki  = (uint16_t*)(ws + o);                               // 8 MB

    int nst = TQ_ * DM_;
    pack_states<<<(nst + 255) / 256, 256, 0, stream>>>(q, qb);
    pack_states<<<(nst + 255) / 256, 256, 0, stream>>>(k, kb);
    int nw = NI_ * DM_;
    pack_weight<<<(nw + 255) / 256, 256, 0, stream>>>(Wq, wqb);
    pack_weight<<<(nw + 255) / 256, 256, 0, stream>>>(Wk, wkb);

    proj_gemm<<<4096, 128, 0, stream>>>(qb, wqb, qi);
    proj_gemm<<<4096, 128, 0, stream>>>(kb, wkb, ki);

    int*   out_idx = (int*)d_out;
    float* out_val = (float*)d_out + (size_t)H_ * TQ_ * NSEL_;

    hipFuncSetAttribute((const void*)score_topk,
                        hipFuncAttributeMaxDynamicSharedMemorySize, (int)LDS_TOTAL);
    score_topk<<<H_ * 256, 128, LDS_TOTAL, stream>>>(qi, ki, mask, out_idx, out_val);
}
